// Efficient_PAM_Module_21827023798785
// MI455X (gfx1250) — matrix operand order repaired and checked by reading, not compiled, not run on hardware
//
#include <hip/hip_runtime.h>
#include <math.h>

typedef __attribute__((ext_vector_type(16))) _Float16 v16h;
typedef __attribute__((ext_vector_type(8)))  float    v8f;
typedef unsigned int u32x4 __attribute__((ext_vector_type(4)));
typedef int          i32x8 __attribute__((ext_vector_type(8)));
typedef int          i32x4 __attribute__((ext_vector_type(4)));
typedef _Float16 half_t;

#define C_DIM   576
#define CQ      64
#define N_DIM   16384
#define H_DIM   128
#define S_SLOTS 48
#define B_DIM   8
#define NT      128          // pixels per workgroup
#define KC      32           // channels per streamed chunk
#define RATIO_  2

union Frag16 { unsigned u[8]; v16h h; };
union FragH  { _Float16 e[16]; v16h h; };

// ---------------------------------------------------------------------------
// WMMA fragment builders (CDNA5 wave32 layouts, CDNA5 ISA §7.12.2)
// ---------------------------------------------------------------------------

// A fragment (16x32 f16) from an f16 buffer row-major [rows][stride],
// K pairs contiguous along the row (one dword per VGPR).
__device__ inline v16h load_a_frag(const half_t* base, int rowbase, int stride,
                                   int kbase, int lane) {
    Frag16 f;
    const int m  = lane & 15;
    const int hf = lane >> 4;
    const unsigned* p =
        (const unsigned*)(base + (size_t)(rowbase + m) * stride + kbase);
#pragma unroll
    for (int v = 0; v < 8; ++v) {
        const int koff = ((v < 4) ? 0 : 16) + hf * 8 + (v & 3) * 2;
        f.u[v] = p[koff >> 1];
    }
    return f.h;
}

// A fragment (16x32 f16) from TDM-landed f32 LDS tile laid out [K=ch][NT=px];
// convert f32->f16 at fragment build (x stream stays raw through the DMA).
__device__ inline v16h load_a_frag_xT(const float* base, int rowbase, int lane) {
    FragH f;
    const int m  = lane & 15;
    const int hf = lane >> 4;
    const float* p = base + rowbase + m;
#pragma unroll
    for (int v = 0; v < 8; ++v) {
        const int k0 = ((v < 4) ? 0 : 16) + hf * 8 + (v & 3) * 2;
        f.e[2 * v]     = (half_t)p[(size_t)k0 * NT];
        f.e[2 * v + 1] = (half_t)p[(size_t)(k0 + 1) * NT];
    }
    return f.h;
}

// B fragment (32x16 KxN f16) from buffer [N][K] (K pair per dword).
__device__ inline v16h load_b_frag(const half_t* base, int nbase, int kstride,
                                   int kbase, int lane) {
    Frag16 f;
    const int n  = nbase + (lane & 15);
    const int hf = lane >> 4;
    const unsigned* p =
        (const unsigned*)(base + (size_t)n * kstride + kbase);
#pragma unroll
    for (int v = 0; v < 8; ++v) f.u[v] = p[((v < 4) ? 0 : 8) + hf * 4 + (v & 3)];
    return f.h;
}

// ---------------------------------------------------------------------------
// TDM: DMA a 2-D f32 tile (KC rows x NT cols, row stride N_DIM) into LDS.
// D# per CDNA5 ISA: group0 = {flags, lds_addr, gaddr,
// gaddr_hi|type}, group1 = dims/tile/strides. Groups 2/3/4 unused (2-D).
// This toolchain's builtin is the 6-arg flavor (g0,g1,g2,g3,g4,cpol).
// ---------------------------------------------------------------------------
__device__ inline void tdm_load_tile(unsigned lds_off, const float* gptr) {
    const unsigned long long ga = (unsigned long long)(const void*)gptr;
    u32x4 g0;
    g0[0] = 1u;                                   // count=1, user mode
    g0[1] = lds_off;                              // LDS byte address
    g0[2] = (unsigned)ga;                         // global_addr[31:0]
    g0[3] = (unsigned)((ga >> 32) & 0x01FFFFFFu)  // global_addr[56:32]
          | (2u << 30);                           // type = 2 ("image")
    i32x8 g1;
    g1[0] = (int)(2u << 16);                      // data_size = 4 bytes
    g1[1] = (int)((unsigned)(N_DIM & 0xFFFF) << 16);   // tensor_dim0 lo
    g1[2] = (int)((N_DIM >> 16) | ((unsigned)C_DIM << 16)); // dim0 hi|dim1 lo
    g1[3] = (int)((unsigned)NT << 16);            // dim1 hi=0 | tile_dim0=128
    g1[4] = (int)KC;                              // tile_dim1=32, tile_dim2=0
    g1[5] = (int)N_DIM;                           // tensor_dim0_stride lo
    g1[6] = 0;                                    // stride0 hi | stride1 lo
    g1[7] = 0;                                    // stride1 hi
    const i32x4 gz4 = {0, 0, 0, 0};
    const i32x8 gz8 = {0, 0, 0, 0, 0, 0, 0, 0};
    __builtin_amdgcn_tensor_load_to_lds(g0, g1, gz4, gz4, gz8, 0);
}

// ---------------------------------------------------------------------------
// Kernel 1: per-batch slot compaction (P == S == 48, trivial serial).
// ---------------------------------------------------------------------------
__global__ void pam_select(const long long* __restrict__ points,
                           int* __restrict__ gidx, int* __restrict__ nvalid) {
    const int b = blockIdx.x;
    if (threadIdx.x != 0) return;
    int cnt = 0;
    for (int j = 0; j < S_SLOTS; ++j) {
        const long long p0 = points[((size_t)b * S_SLOTS + j) * 2 + 0];
        const long long p1 = points[((size_t)b * S_SLOTS + j) * 2 + 1];
        const bool inr = (p0 < H_DIM * RATIO_) && (p1 < H_DIM * RATIO_);
        const long long sel = (p0 / RATIO_) * H_DIM + (p1 / RATIO_);
        if (inr && sel > 0 && cnt < S_SLOTS) {
            gidx[b * S_SLOTS + cnt] = (int)sel;
            ++cnt;
        }
    }
    nvalid[b] = cnt;
}

// ---------------------------------------------------------------------------
// Kernel 2: Wq f32[64][576] -> f16 same layout ([N=o][K=c]: B-operand ready).
// ---------------------------------------------------------------------------
__global__ void pam_cvt_wq(const float* __restrict__ Wq,
                           half_t* __restrict__ wqh) {
    const int i = blockIdx.x * 256 + threadIdx.x;
    if (i < CQ * C_DIM) wqh[i] = (half_t)Wq[i];
}

// ---------------------------------------------------------------------------
// Kernel 3: gather selected x columns, project with Wk/Wv.
//   ksel[b][s][o]   f16 (energy B-operand [N=s][K=o])
//   vsel[b][c][s64] f16 (out    B-operand [N=c][K=s], s padded to 64)
// grid = (10 channel tiles, B): tile 0 -> 64 k-channels, tiles 1..9 -> v.
// ---------------------------------------------------------------------------
__global__ __launch_bounds__(256) void pam_prep(
    const float* __restrict__ x, const float* __restrict__ Wk,
    const float* __restrict__ bk, const float* __restrict__ Wv,
    const float* __restrict__ bv, const int* __restrict__ gidx,
    const int* __restrict__ nvalid, half_t* __restrict__ ksel,
    half_t* __restrict__ vsel) {
    __shared__ half_t xg[S_SLOTS * C_DIM];   // 55 KB
    const int t = threadIdx.x;
    const int b = blockIdx.y;
    const int tile = blockIdx.x;
    const int nv = nvalid[b];
    const size_t xbase = (size_t)b * C_DIM * N_DIM;

    for (int idx = t; idx < S_SLOTS * C_DIM; idx += 256) {
        const int s = idx / C_DIM, c = idx % C_DIM;
        half_t val = (half_t)0.f;
        if (s < nv) {
            const int g = gidx[b * S_SLOTS + s];
            val = (half_t)x[xbase + (size_t)c * N_DIM + g];
        }
        xg[idx] = val;
    }
    __syncthreads();

    const int ocb = tile * 64;
    for (int j = t; j < 64 * S_SLOTS; j += 256) {
        const int ol = j / S_SLOTS, s = j % S_SLOTS;
        const int oc = ocb + ol;
        float acc = 0.f;
        if (s < nv) {
            const float* wrow;
            if (oc < CQ) { wrow = Wk + (size_t)oc * C_DIM; acc = bk[oc]; }
            else         { wrow = Wv + (size_t)(oc - CQ) * C_DIM; acc = bv[oc - CQ]; }
            for (int c = 0; c < C_DIM; ++c)
                acc += (float)xg[s * C_DIM + c] * wrow[c];
        }
        if (oc < CQ)
            ksel[(size_t)b * S_SLOTS * 64 + (size_t)s * 64 + oc] = (half_t)acc;
        else
            vsel[(size_t)b * C_DIM * 64 + (size_t)(oc - CQ) * 64 + s] = (half_t)acc;
    }
    if (ocb >= CQ) {  // zero K padding s=48..63 for this tile's v channels
        for (int j = t; j < 64 * 16; j += 256) {
            const int ol = j / 16, s = S_SLOTS + (j % 16);
            vsel[(size_t)b * C_DIM * 64 + (size_t)(ocb - CQ + ol) * 64 + s] =
                (half_t)0.f;
        }
    }
}

// ---------------------------------------------------------------------------
// Kernel 4: fused q-GEMM -> energy -> softmax -> out-GEMM + residual.
// One workgroup = batch b x 128-pixel tile, 8 waves, one 16-px M-row each.
// x is streamed by the Tensor Data Mover into double-buffered LDS tiles.
// ---------------------------------------------------------------------------
__global__ __launch_bounds__(256) void pam_main(
    const float* __restrict__ x, const half_t* __restrict__ wqh,
    const float* __restrict__ bq, const half_t* __restrict__ ksel,
    const half_t* __restrict__ vsel, const int* __restrict__ nvalid,
    const float* __restrict__ gamma, float* __restrict__ out) {
    // Aliased LDS (phases are barrier-separated):
    //  [0,16K)  xbuf0 f32 [32][128]   phase 1 (TDM dst, chunk even)
    //  [16,32K) xbuf1 f32 [32][128]   phase 1 (TDM dst, chunk odd)
    //  [0,24K)  eM    f32 [128][48]   phase 2/3 (aliases xbuf0 + half xbuf1)
    //  [32,48K) qh    f16 [128][64]   phase 1.5/2; ah aliases it in 3/4
    __shared__ __align__(16) unsigned char smem[49152];
    float*  xb0 = (float*)smem;
    float*  xb1 = (float*)(smem + 16384);
    float*  eM  = (float*)smem;
    half_t* qh  = (half_t*)(smem + 32768);
    half_t* ah  = qh;

    const int t    = threadIdx.x;
    const int lane = t & 31;
    const int wave = t >> 5;                       // 0..7
    const int b    = blockIdx.y;
    const int n0   = blockIdx.x * NT;
    const int nv   = nvalid[b];
    const float g  = gamma[0];
    const size_t xbase = (size_t)b * C_DIM * N_DIM;
    const int nlo  = lane & 15;
    const int hf   = lane >> 4;
    const unsigned lds0 = (unsigned)(uintptr_t)(void*)smem;

    // -------- Phase 1: q = x^T * Wq^T  (M=128 px, N=64, K=576) --------
    v8f qacc[4];
#pragma unroll
    for (int ti = 0; ti < 4; ++ti)
#pragma unroll
        for (int v = 0; v < 8; ++v) qacc[ti][v] = 0.f;

    const int NKC = C_DIM / KC;                    // 18 chunks
    if (wave == 0)                                 // prime the pipeline
        tdm_load_tile(lds0, x + xbase + n0);

    for (int kc = 0; kc < NKC; ++kc) {
        if (wave == 0) {
            if (kc + 1 < NKC) {                    // prefetch next chunk
                tdm_load_tile(lds0 + (unsigned)(((kc + 1) & 1) * 16384),
                              x + xbase + (size_t)(kc + 1) * KC * N_DIM + n0);
                __builtin_amdgcn_s_wait_tensorcnt(1);  // chunk kc landed
            } else {
                __builtin_amdgcn_s_wait_tensorcnt(0);
            }
        }
        __syncthreads();                           // LDS tile visible to all
        const float* cur = (kc & 1) ? xb1 : xb0;
        const v16h a = load_a_frag_xT(cur, wave * 16, lane);
#pragma unroll
        for (int ti = 0; ti < 4; ++ti) {
            const v16h bf = load_b_frag(wqh, ti * 16, C_DIM, kc * KC, lane);
            qacc[ti] = __builtin_amdgcn_wmma_f32_16x16x32_f16(
                false, a, false, bf, (short)0, qacc[ti], false, false);
        }
        __syncthreads();                           // before buffer reuse
    }

    // bias + energy scale (sqrt(9/576) = 0.125 folded into q), spill to LDS
#pragma unroll
    for (int ti = 0; ti < 4; ++ti) {
        const float bqv = bq[ti * 16 + nlo];
#pragma unroll
        for (int v = 0; v < 8; ++v) {
            const int px = wave * 16 + v + 8 * hf;
            qh[px * 64 + ti * 16 + nlo] =
                (half_t)((qacc[ti][v] + bqv) * 0.125f);
        }
    }
    __syncthreads();

    // -------- Phase 2: energy = q * k_sel  (N=48 slots, K=64) --------
    v8f efr[3];
#pragma unroll
    for (int ti = 0; ti < 3; ++ti)
#pragma unroll
        for (int v = 0; v < 8; ++v) efr[ti][v] = 0.f;
    const half_t* kb = ksel + (size_t)b * S_SLOTS * 64;
#pragma unroll
    for (int ks = 0; ks < 2; ++ks) {
        const v16h a = load_a_frag(qh, wave * 16, 64, ks * 32, lane);
#pragma unroll
        for (int ti = 0; ti < 3; ++ti) {
            const v16h bf = load_b_frag(kb, ti * 16, 64, ks * 32, lane);
            efr[ti] = __builtin_amdgcn_wmma_f32_16x16x32_f16(
                false, a, false, bf, (short)0, efr[ti], false, false);
        }
    }
#pragma unroll
    for (int ti = 0; ti < 3; ++ti)
#pragma unroll
        for (int v = 0; v < 8; ++v) {
            const int px = wave * 16 + v + 8 * hf;
            eM[px * S_SLOTS + ti * 16 + nlo] = efr[ti][v];
        }
    __syncthreads();

    // -------- Phase 3: masked softmax over slots, attn f16 (K padded 64) ----
    if (t < NT) {
        const int px = t;
        if (nv <= 0) {
            for (int s = 0; s < 64; ++s) ah[px * 64 + s] = (half_t)0.f;
        } else {
            float mx = -3.0e38f;
            for (int s = 0; s < nv; ++s) mx = fmaxf(mx, eM[px * S_SLOTS + s]);
            float sum = 0.f;
            for (int s = 0; s < nv; ++s) sum += expf(eM[px * S_SLOTS + s] - mx);
            const float inv = 1.f / fmaxf(sum, 1e-30f);
            for (int s = 0; s < 64; ++s)
                ah[px * 64 + s] = (s < nv)
                    ? (half_t)(expf(eM[px * S_SLOTS + s] - mx) * inv)
                    : (half_t)0.f;
        }
    }
    __syncthreads();

    // -------- Phase 4: out = gamma * (v_sel * attn^T) + x  (N=576, K=64) ----
    const v16h a0 = load_a_frag(ah, wave * 16, 64, 0, lane);
    const v16h a1 = load_a_frag(ah, wave * 16, 64, 32, lane);
    const half_t* vb = vsel + (size_t)b * C_DIM * 64;
    for (int ct = 0; ct < C_DIM / 16; ++ct) {      // 36 channel tiles
        v8f o;
#pragma unroll
        for (int v = 0; v < 8; ++v) o[v] = 0.f;
        o = __builtin_amdgcn_wmma_f32_16x16x32_f16(
            false, a0, false, load_b_frag(vb, ct * 16, 64, 0, lane),
            (short)0, o, false, false);
        o = __builtin_amdgcn_wmma_f32_16x16x32_f16(
            false, a1, false, load_b_frag(vb, ct * 16, 64, 32, lane),
            (short)0, o, false, false);
        const int c = ct * 16 + nlo;
        const size_t base =
            xbase + (size_t)c * N_DIM + n0 + wave * 16 + 8 * hf;
        // 8 consecutive pixels, 32B aligned -> two b128 loads/stores
        const float4 x0 = *(const float4*)(x + base);
        const float4 x1 = *(const float4*)(x + base + 4);
        float4 r0, r1;
        r0.x = g * o[0] + x0.x; r0.y = g * o[1] + x0.y;
        r0.z = g * o[2] + x0.z; r0.w = g * o[3] + x0.w;
        r1.x = g * o[4] + x1.x; r1.y = g * o[5] + x1.y;
        r1.z = g * o[6] + x1.z; r1.w = g * o[7] + x1.w;
        *(float4*)(out + base)     = r0;
        *(float4*)(out + base + 4) = r1;
    }
}

// ---------------------------------------------------------------------------
extern "C" void kernel_launch(void* const* d_in, const int* in_sizes, int n_in,
                              void* d_out, int out_size, void* d_ws,
                              size_t ws_size, hipStream_t stream) {
    const float*     x      = (const float*)d_in[0];
    const long long* points = (const long long*)d_in[1];
    const float*     Wq     = (const float*)d_in[2];
    const float*     bq     = (const float*)d_in[3];
    const float*     Wk     = (const float*)d_in[4];
    const float*     bk     = (const float*)d_in[5];
    const float*     Wv     = (const float*)d_in[6];
    const float*     bv     = (const float*)d_in[7];
    const float*     gamma  = (const float*)d_in[8];
    float* out = (float*)d_out;

    char* ws = (char*)d_ws;
    int*    gidx   = (int*)ws;                                   // 1536 B
    int*    nvalid = (int*)(ws + 2048);                          // 32 B
    half_t* wqh    = (half_t*)(ws + 4096);                       // 73728 B
    half_t* ksel   = (half_t*)(ws + 4096 + 73728);               // 49152 B
    half_t* vsel   = (half_t*)(ws + 4096 + 73728 + 49152);       // 589824 B

    pam_select<<<B_DIM, 32, 0, stream>>>(points, gidx, nvalid);
    pam_cvt_wq<<<(CQ * C_DIM + 255) / 256, 256, 0, stream>>>(Wq, wqh);
    pam_prep<<<dim3(10, B_DIM), 256, 0, stream>>>(x, Wk, bk, Wv, bv, gidx,
                                                  nvalid, ksel, vsel);
    pam_main<<<dim3(N_DIM / NT, B_DIM), 256, 0, stream>>>(
        x, wqh, bq, ksel, vsel, nvalid, gamma, out);
}
